// MultiHeadAttention_55052890800896
// MI455X (gfx1250) — compile-verified, matrix operand order checked by reading, not run on hardware
//
#include <hip/hip_runtime.h>
#include <hip/hip_bf16.h>

typedef __attribute__((ext_vector_type(16))) __bf16 v16bf;
typedef __attribute__((ext_vector_type(8)))  __bf16 v8bf;
typedef __attribute__((ext_vector_type(4)))  __bf16 v4bf;
typedef __attribute__((ext_vector_type(8)))  float  v8f;

#define D_MODEL 1024
#define N_HEAD  16
#define D_K     64
#define SEQ     2048
#define BATCH   2
#define BH      (BATCH * N_HEAD)     // 32
#define MROWS   (BATCH * SEQ)        // 4096

// ---------------------------------------------------------------------------
// helpers
// ---------------------------------------------------------------------------
__device__ __forceinline__ v8f wmma_bf16(v16bf a, v16bf b, v8f c) {
  // V_WMMA_F32_16X16X32_BF16 : D = A(16x32) * B(32x16) + C(16x16 f32)
  return __builtin_amdgcn_wmma_f32_16x16x32_bf16(
      /*neg_a=*/false, a, /*neg_b=*/false, b,
      /*c_mod=*/(short)0, c, /*reuse_a=*/false, /*reuse_b=*/false);
}

// Load one 16x32 bf16 operand fragment from an LDS tile laid out [row][k],
// row-stride `ld` elements (multiple of 8 -> every v8bf read is 16B aligned).
// ISA 16-bit A layout: lanes 0-15 -> row=lane, K {0..7,16..23};
// lanes 16-31 -> row=lane-16, K {8..15,24..31}. B operand mirrors with row=N.
__device__ __forceinline__ v16bf frag_ld(const __bf16* tile, int row, int k0,
                                         int ld, int lane) {
  const int r  = row + (lane & 15);
  const int kk = k0 + ((lane >> 4) << 3);
  v8bf lo = *(const v8bf*)(tile + r * ld + kk);
  v8bf hi = *(const v8bf*)(tile + r * ld + kk + 16);
  v16bf out;
#pragma unroll
  for (int i = 0; i < 8; ++i) { out[i] = lo[i]; out[i + 8] = hi[i]; }
  return out;
}

// Async DMA: global (16B) -> LDS, tracked by ASYNCcnt (no VGPR data path).
__device__ __forceinline__ void async_ld_b128(unsigned lds_off,
                                              unsigned long long gaddr) {
  asm volatile("global_load_async_to_lds_b128 %0, %1, off"
               :
               : "v"(lds_off), "v"(gaddr)
               : "memory");
}
__device__ __forceinline__ void wait_async0() {
  asm volatile("s_wait_asynccnt 0x0" ::: "memory");
}

// XOR-butterfly reductions across 16-lane halves via ds_swizzle_b32
// (group-of-32 encoding: xor_mask<<10 | and_mask(0x1f)).
__device__ __forceinline__ float red16_max(float x) {
  x = fmaxf(x, __int_as_float(__builtin_amdgcn_ds_swizzle(__float_as_int(x), 0x041f)));
  x = fmaxf(x, __int_as_float(__builtin_amdgcn_ds_swizzle(__float_as_int(x), 0x081f)));
  x = fmaxf(x, __int_as_float(__builtin_amdgcn_ds_swizzle(__float_as_int(x), 0x101f)));
  x = fmaxf(x, __int_as_float(__builtin_amdgcn_ds_swizzle(__float_as_int(x), 0x201f)));
  return x;
}
__device__ __forceinline__ float red16_sum(float x) {
  x += __int_as_float(__builtin_amdgcn_ds_swizzle(__float_as_int(x), 0x041f));
  x += __int_as_float(__builtin_amdgcn_ds_swizzle(__float_as_int(x), 0x081f));
  x += __int_as_float(__builtin_amdgcn_ds_swizzle(__float_as_int(x), 0x101f));
  x += __int_as_float(__builtin_amdgcn_ds_swizzle(__float_as_int(x), 0x201f));
  return x;
}

// ---------------------------------------------------------------------------
// fp32 -> bf16 conversion (memory-bound prologue)
// ---------------------------------------------------------------------------
__global__ __launch_bounds__(256) void cvt_f32_bf16(const float* __restrict__ x,
                                                    __bf16* __restrict__ y,
                                                    int n4) {
  int i = blockIdx.x * blockDim.x + threadIdx.x;
  if (i < n4) {
    const float4 v = ((const float4*)x)[i];
    v4bf o;
    o[0] = (__bf16)v.x; o[1] = (__bf16)v.y;
    o[2] = (__bf16)v.z; o[3] = (__bf16)v.w;
    ((v4bf*)y)[i] = o;
  }
}

// ---------------------------------------------------------------------------
// NT GEMM: Y[m,n] = sum_k A[m,k]*B[n,k] + bias[n]
//   A: [M][K] bf16 row-major, B: [N][K] bf16 row-major (torch Linear weight)
//   OUT_MODE 0: bf16 output scattered to head-split [B,H,S,Dk] layout
//   OUT_MODE 1: f32 output row-major [M][N]
// 256 threads (8 waves, 2x4), tile 128x128, K-step 32.
// Double-buffered LDS fed by GLOBAL_LOAD_ASYNC_TO_LDS_B128 (ASYNCcnt).
// ---------------------------------------------------------------------------
#define GT 5120                       // elems per 128x40 tile
template <int OUT_MODE>
__global__ __launch_bounds__(256) void gemm_nt(const __bf16* __restrict__ A,
                                               const __bf16* __restrict__ B,
                                               const float* __restrict__ bias,
                                               void* __restrict__ Out,
                                               int M, int N, int K) {
  __shared__ __align__(16) __bf16 smem[4 * GT];  // [buf][A|B] -> 40 KiB

  const int tid   = threadIdx.x;
  const int lane  = tid & 31;
  const int w     = tid >> 5;   // wave 0..7
  const int wm    = w >> 2;     // 0..1  (64 rows each)
  const int wn    = w & 3;      // 0..3  (32 cols each)
  const int mBase = blockIdx.y * 128;
  const int nBase = blockIdx.x * 128;

  // stage one 128x32 A tile + B tile into buffer `b` via async DMA
  auto stage = [&](int b, int k0) {
    const unsigned aoff = (unsigned)(b * 2 * GT) * 2u;      // bytes
    const unsigned boff = aoff + (unsigned)GT * 2u;
#pragma unroll
    for (int i = 0; i < 2; ++i) {
      const int c   = tid + i * 256;    // 512 chunks of 8 bf16
      const int row = c >> 2;
      const int off = (c & 3) << 3;
      const unsigned l = (unsigned)(row * 40 + off) * 2u;
      async_ld_b128(aoff + l,
                    (unsigned long long)(A + (size_t)(mBase + row) * K + k0 + off));
      async_ld_b128(boff + l,
                    (unsigned long long)(B + (size_t)(nBase + row) * K + k0 + off));
    }
  };

  v8f acc[4][2];
#pragma unroll
  for (int at = 0; at < 4; ++at)
#pragma unroll
    for (int bt = 0; bt < 2; ++bt) {
      v8f z = {0.f, 0.f, 0.f, 0.f, 0.f, 0.f, 0.f, 0.f};
      acc[at][bt] = z;
    }

  const int NK = K >> 5;
  int buf = 0;
  stage(0, 0);
  wait_async0();
  __syncthreads();

  for (int kb = 0; kb < NK; ++kb) {
    if (kb + 1 < NK) stage(buf ^ 1, (kb + 1) << 5);  // DMA next tile, overlap

    const __bf16* As = smem + buf * 2 * GT;
    const __bf16* Bs = As + GT;
    v16bf bfrag[2];
#pragma unroll
    for (int bt = 0; bt < 2; ++bt)
      bfrag[bt] = frag_ld(Bs, wn * 32 + bt * 16, 0, 40, lane);
#pragma unroll
    for (int at = 0; at < 4; ++at) {
      v16bf afrag = frag_ld(As, wm * 64 + at * 16, 0, 40, lane);
#pragma unroll
      for (int bt = 0; bt < 2; ++bt)
        acc[at][bt] = wmma_bf16(afrag, bfrag[bt], acc[at][bt]);
    }

    wait_async0();
    __syncthreads();
    buf ^= 1;
  }

  // epilogue: C/D layout -> reg j holds row j (lanes 0-15) / j+8 (lanes 16-31)
  const int rhi = (lane >> 4) << 3;
#pragma unroll
  for (int at = 0; at < 4; ++at)
#pragma unroll
    for (int bt = 0; bt < 2; ++bt) {
      const int col = nBase + wn * 32 + bt * 16 + (lane & 15);
      const float bv = bias[col];
#pragma unroll
      for (int j = 0; j < 8; ++j) {
        const int row = mBase + wm * 64 + at * 16 + rhi + j;
        const float val = acc[at][bt][j] + bv;
        if (OUT_MODE == 0) {
          const int b  = row >> 11;          // row / SEQ
          const int s  = row & (SEQ - 1);
          const int h  = col >> 6;           // col / D_K
          const int dk = col & (D_K - 1);
          ((__bf16*)Out)[((size_t)((b << 4) + h) * SEQ + s) * D_K + dk] =
              (__bf16)val;
        } else {
          ((float*)Out)[(size_t)row * N + col] = val;
        }
      }
    }
}

// ---------------------------------------------------------------------------
// Flash attention per (b,h): Q block 128 rows/WG, 64-key inner blocks.
// Qh/Kh/Vh: [BH][SEQ][D_K] bf16.  AO: [MROWS][D_MODEL] bf16 (merged heads).
// Q and K tiles staged by async DMA; V transposed through VGPRs.
// ---------------------------------------------------------------------------
#define QS_ELE (128 * 72)             // 9216
#define VT_ELE (64 * 72)              // 4608
#define KP_ELE (8 * 16 * 72)          // 9216
__global__ __launch_bounds__(256) void attn_flash(const __bf16* __restrict__ Qh,
                                                  const __bf16* __restrict__ Kh,
                                                  const __bf16* __restrict__ Vh,
                                                  __bf16* __restrict__ AO) {
  __shared__ __align__(16) __bf16 smem[QS_ELE + VT_ELE + KP_ELE];  // 45 KiB
  __bf16* Qs = smem;                  // [qrow][d]
  __bf16* Vt = smem + QS_ELE;         // [dk][key] (transposed)
  __bf16* KP = smem + QS_ELE + VT_ELE;  // K tile [key][d] / per-wave P tiles
  const unsigned KP_BOFF = (unsigned)(QS_ELE + VT_ELE) * 2u;

  const int tid  = threadIdx.x;
  const int lane = tid & 31;
  const int w    = tid >> 5;          // wave -> q rows [w*16, w*16+16)
  const int bh   = blockIdx.y;        // 0..31
  const int qb   = blockIdx.x;        // 0..15
  const size_t base = (size_t)bh * SEQ * D_K;

  // async-stage Q block 128x64 (1024 chunks of 8 bf16; 4 per thread)
#pragma unroll
  for (int i = 0; i < 4; ++i) {
    const int c   = tid + i * 256;
    const int row = c >> 3;
    const int off = (c & 7) << 3;
    async_ld_b128((unsigned)(row * 72 + off) * 2u,
                  (unsigned long long)(Qh + base +
                                       (size_t)(qb * 128 + row) * D_K + off));
  }

  float mrow[8], lrow[8];
  v8f o[4];
#pragma unroll
  for (int j = 0; j < 8; ++j) { mrow[j] = -1e30f; lrow[j] = 0.f; }
#pragma unroll
  for (int nt = 0; nt < 4; ++nt) {
    v8f z = {0.f, 0.f, 0.f, 0.f, 0.f, 0.f, 0.f, 0.f};
    o[nt] = z;
  }
  const float scale = 0.125f;  // 1/sqrt(D_K)

  for (int kb = 0; kb < SEQ / 64; ++kb) {
    __syncthreads();  // all waves done with previous K/P region
    // K tile 64x64 -> KP rows 0..63 ([key][d]) via async DMA
#pragma unroll
    for (int i = 0; i < 2; ++i) {
      const int c   = tid + i * 256;
      const int row = c >> 3;
      const int off = (c & 7) << 3;
      async_ld_b128(KP_BOFF + (unsigned)(row * 72 + off) * 2u,
                    (unsigned long long)(Kh + base +
                                         (size_t)(kb * 64 + row) * D_K + off));
    }
    // V tile transposed through VGPRs: Vt[dk][key]
#pragma unroll
    for (int i = 0; i < 16; ++i) {
      const int idx = tid + i * 256;  // 4096 elements, coalesced reads
      const int key = idx >> 6;
      const int dk  = idx & 63;
      Vt[dk * 72 + key] = Vh[base + (size_t)(kb * 64 + key) * D_K + dk];
    }
    wait_async0();
    __syncthreads();

    // S = Q * K^T : 16(q) x 64(key) per wave, k-dim = d = 64 (2 WMMA steps)
    v8f sc[4];
#pragma unroll
    for (int nt = 0; nt < 4; ++nt) {
      v8f z = {0.f, 0.f, 0.f, 0.f, 0.f, 0.f, 0.f, 0.f};
      sc[nt] = z;
    }
#pragma unroll
    for (int ks = 0; ks < 2; ++ks) {
      v16bf qa = frag_ld(Qs, w * 16, ks * 32, 72, lane);
#pragma unroll
      for (int nt = 0; nt < 4; ++nt) {
        v16bf kf = frag_ld(KP, nt * 16, ks * 32, 72, lane);
        sc[nt] = wmma_bf16(qa, kf, sc[nt]);
      }
    }

    // online softmax (row stats shared across 16-lane halves via ds_swizzle)
#pragma unroll
    for (int j = 0; j < 8; ++j) {
      float mx = -1e30f;
#pragma unroll
      for (int nt = 0; nt < 4; ++nt) mx = fmaxf(mx, sc[nt][j] * scale);
      mx = red16_max(mx);
      const float nm    = fmaxf(mrow[j], mx);
      const float alpha = __expf(mrow[j] - nm);
      mrow[j] = nm;
      float rs = 0.f;
#pragma unroll
      for (int nt = 0; nt < 4; ++nt) {
        const float p = __expf(sc[nt][j] * scale - nm);
        sc[nt][j] = p;
        rs += p;
      }
      rs = red16_sum(rs);
      lrow[j] = lrow[j] * alpha + rs;
#pragma unroll
      for (int nt = 0; nt < 4; ++nt) o[nt][j] *= alpha;
    }

    __syncthreads();  // all waves done reading K tile before P overwrites it

    // write P (bf16) into this wave's private LDS region (A-operand source)
    __bf16* Ps = KP + w * 16 * 72;
    {
      const int rhi = (lane >> 4) << 3;
#pragma unroll
      for (int nt = 0; nt < 4; ++nt)
#pragma unroll
        for (int j = 0; j < 8; ++j)
          Ps[(rhi + j) * 72 + nt * 16 + (lane & 15)] = (__bf16)sc[nt][j];
    }

    // O += P * V : A = P (16x64), B = Vt -> 16 x 64(dk)
#pragma unroll
    for (int ks = 0; ks < 2; ++ks) {
      v16bf pa = frag_ld(Ps, 0, ks * 32, 72, lane);
#pragma unroll
      for (int nt = 0; nt < 4; ++nt) {
        v16bf vf = frag_ld(Vt, nt * 16, ks * 32, 72, lane);
        o[nt] = wmma_bf16(pa, vf, o[nt]);
      }
    }
  }

  // epilogue: normalize and write merged-head bf16 activations
  const int b = bh >> 4;
  const int h = bh & 15;
  const int rhi = (lane >> 4) << 3;
#pragma unroll
  for (int j = 0; j < 8; ++j) {
    const float inv = 1.0f / lrow[j];
    const int m = b * SEQ + qb * 128 + w * 16 + rhi + j;
#pragma unroll
    for (int nt = 0; nt < 4; ++nt) {
      const int col = h * D_K + nt * 16 + (lane & 15);
      AO[(size_t)m * D_MODEL + col] = (__bf16)(o[nt][j] * inv);
    }
  }
}

// ---------------------------------------------------------------------------
// launcher
// ---------------------------------------------------------------------------
extern "C" void kernel_launch(void* const* d_in, const int* in_sizes, int n_in,
                              void* d_out, int out_size, void* d_ws,
                              size_t ws_size, hipStream_t stream) {
  (void)in_sizes; (void)n_in; (void)out_size; (void)ws_size;
  const float* q  = (const float*)d_in[0];
  const float* k  = (const float*)d_in[1];
  const float* v  = (const float*)d_in[2];
  const float* Wq = (const float*)d_in[3];
  const float* bq = (const float*)d_in[4];
  const float* Wk = (const float*)d_in[5];
  const float* bk = (const float*)d_in[6];
  const float* Wv = (const float*)d_in[7];
  const float* bv = (const float*)d_in[8];
  const float* Wo = (const float*)d_in[9];
  const float* bo = (const float*)d_in[10];

  const size_t szX = (size_t)MROWS * D_MODEL * sizeof(__bf16);    // 8 MiB
  const size_t szW = (size_t)D_MODEL * D_MODEL * sizeof(__bf16);  // 2 MiB
  char* ws = (char*)d_ws;
  __bf16* qb16 = (__bf16*)ws; ws += szX;
  __bf16* kb16 = (__bf16*)ws; ws += szX;
  __bf16* vb16 = (__bf16*)ws; ws += szX;
  __bf16* wq16 = (__bf16*)ws; ws += szW;
  __bf16* wk16 = (__bf16*)ws; ws += szW;
  __bf16* wv16 = (__bf16*)ws; ws += szW;
  __bf16* wo16 = (__bf16*)ws; ws += szW;
  __bf16* Qhd  = (__bf16*)ws; ws += szX;  // [BH][SEQ][D_K]
  __bf16* Khd  = (__bf16*)ws; ws += szX;
  __bf16* Vhd  = (__bf16*)ws; ws += szX;
  __bf16* AO   = (__bf16*)ws; ws += szX;  // merged-head attention output

  const int nX4 = (MROWS * D_MODEL) / 4;    // 1,048,576
  const int nW4 = (D_MODEL * D_MODEL) / 4;  // 262,144
  cvt_f32_bf16<<<nX4 / 256, 256, 0, stream>>>(q,  qb16, nX4);
  cvt_f32_bf16<<<nX4 / 256, 256, 0, stream>>>(k,  kb16, nX4);
  cvt_f32_bf16<<<nX4 / 256, 256, 0, stream>>>(v,  vb16, nX4);
  cvt_f32_bf16<<<nW4 / 256, 256, 0, stream>>>(Wq, wq16, nW4);
  cvt_f32_bf16<<<nW4 / 256, 256, 0, stream>>>(Wk, wk16, nW4);
  cvt_f32_bf16<<<nW4 / 256, 256, 0, stream>>>(Wv, wv16, nW4);
  cvt_f32_bf16<<<nW4 / 256, 256, 0, stream>>>(Wo, wo16, nW4);

  const dim3 ggrid(D_MODEL / 128, MROWS / 128);  // (8, 32)
  gemm_nt<0><<<ggrid, 256, 0, stream>>>(qb16, wq16, bq, (void*)Qhd,
                                        MROWS, D_MODEL, D_MODEL);
  gemm_nt<0><<<ggrid, 256, 0, stream>>>(kb16, wk16, bk, (void*)Khd,
                                        MROWS, D_MODEL, D_MODEL);
  gemm_nt<0><<<ggrid, 256, 0, stream>>>(vb16, wv16, bv, (void*)Vhd,
                                        MROWS, D_MODEL, D_MODEL);

  attn_flash<<<dim3(SEQ / 128, BH), 256, 0, stream>>>(Qhd, Khd, Vhd, AO);

  gemm_nt<1><<<ggrid, 256, 0, stream>>>(AO, wo16, bo, d_out,
                                        MROWS, D_MODEL, D_MODEL);
}
